// GraphAttentionLayer_41317585387821
// MI455X (gfx1250) — compile-verified, matrix operand order checked by reading, not run on hardware
//
#include <hip/hip_runtime.h>
#include <hip/hip_bf16.h>

typedef unsigned int  u32;
typedef unsigned short u16;

typedef __attribute__((ext_vector_type(16))) __bf16 v16bf;
typedef __attribute__((ext_vector_type(8)))  float  v8f;

// 16 bf16 elements of a WMMA A/B fragment, fillable via two 16B vector loads.
union Frag {
    uint4 q[2];
    u16   s[16];
    v16bf v;
};

__device__ __forceinline__ u16 f2bf(float f) {
    u32 u = __builtin_bit_cast(u32, f);
    u += 0x7FFFu + ((u >> 16) & 1u);          // round-to-nearest-even
    return (u16)(u >> 16);
}
__device__ __forceinline__ float bf2f(u16 s) {
    return __builtin_bit_cast(float, ((u32)s) << 16);
}

// ---------------------------------------------------------------------------
// Constants for this problem instance
// ---------------------------------------------------------------------------
#define NB     2
#define NT     4
#define NN     1024
#define FIN    256
#define NH     8
#define ND     128
#define FOUT   1024          // NH*ND
#define SLOPE  0.2f
#define NEGINF (-9e16f)

// ---------------------------------------------------------------------------
// fp32 -> bf16 converters
// ---------------------------------------------------------------------------
__global__ void cvt_w_kernel(const float* __restrict__ W, u16* __restrict__ WT) {
    int idx = blockIdx.x * blockDim.x + threadIdx.x;     // over FOUT*FIN
    if (idx >= FOUT * FIN) return;
    int f = idx >> 8;            // 0..1023
    int s = idx & (FIN - 1);     // 0..255
    WT[(size_t)f * FIN + s] = f2bf(W[(size_t)s * FOUT + f]);   // transpose
}

__global__ void cvt_h_kernel(const float* __restrict__ h, u16* __restrict__ hb, int n) {
    int idx = blockIdx.x * blockDim.x + threadIdx.x;
    if (idx < n) hb[idx] = f2bf(h[idx]);
}

// ---------------------------------------------------------------------------
// GEMM1: ht[bt] = h[bt] (1024x256, bf16) @ W (256x1024, bf16)
// Stores result TRANSPOSED in bf16: htT[bt][f][n]  (K of GEMM2 contiguous).
// Block = 256 threads = 8 waves, computes a 64x128 output tile.
// Each wave: 16 rows x 64 cols = 4 x v_wmma_f32_16x16x32_bf16 per k-step,
// A-fragment reused across the 4 column tiles.
// ---------------------------------------------------------------------------
__global__ __launch_bounds__(256)
void gat_gemm1_kernel(const u16* __restrict__ hb, const u16* __restrict__ WT,
                      u16* __restrict__ htT) {
    const int lane = threadIdx.x & 31;
    const int wave = threadIdx.x >> 5;
    const int wm   = wave & 3;          // row sub-tile
    const int wn   = wave >> 2;         // col group (0/1)
    const int bt   = blockIdx.z;
    const int i0   = blockIdx.y * 64 + wm * 16;
    const int f0   = blockIdx.x * 128 + wn * 64;
    const int m    = lane & 15;         // A row / B col within tile
    const int half = lane >> 4;
    const int kb   = half << 3;         // K sub-block per ISA 16-bit layout

    const u16* Arow = hb + (size_t)bt * (NN * FIN) + (size_t)(i0 + m) * FIN;

    v8f acc[4] = {};
    for (int k0 = 0; k0 < FIN; k0 += 32) {
        Frag a;
        a.q[0] = *(const uint4*)(Arow + k0 + kb);        // K = k0+kb .. +7
        a.q[1] = *(const uint4*)(Arow + k0 + 16 + kb);   // K = k0+16+kb .. +7
#pragma unroll
        for (int t = 0; t < 4; ++t) {
            const u16* Bcol = WT + (size_t)(f0 + t * 16 + m) * FIN;
            Frag b;
            b.q[0] = *(const uint4*)(Bcol + k0 + kb);
            b.q[1] = *(const uint4*)(Bcol + k0 + 16 + kb);
            acc[t] = __builtin_amdgcn_wmma_f32_16x16x32_bf16(
                false, a.v, false, b.v, (short)0, acc[t], false, false);
        }
    }

    // Transposed bf16 store: C/D layout is (M = v + 8*half, N = lane&15).
#pragma unroll
    for (int t = 0; t < 4; ++t) {
        u16* dst = htT + (size_t)bt * (NN * FOUT) +
                   (size_t)(f0 + t * 16 + m) * NN + i0;
#pragma unroll
        for (int v = 0; v < 8; ++v) {
            dst[v + 8 * half] = f2bf(acc[t][v]);
        }
    }
}

// ---------------------------------------------------------------------------
// Scores: src[bt][h][n] = sum_d htT[bt][h*128+d][n] * a[h][d]
//         tgt[bt][h][n] = sum_d htT[bt][h*128+d][n] * a[h][128+d]
// ---------------------------------------------------------------------------
__global__ __launch_bounds__(256)
void gat_scores_kernel(const u16* __restrict__ htT, const float* __restrict__ a,
                       float* __restrict__ srcv, float* __restrict__ tgtv) {
    int idx = blockIdx.x * blockDim.x + threadIdx.x;   // over NB*NT*NH*NN
    int n  = idx & (NN - 1);
    int hh = (idx >> 10) & (NH - 1);
    int bt = idx >> 13;
    const u16*   col = htT + (size_t)bt * (NN * FOUT) + (size_t)(hh * ND) * NN + n;
    const float* as  = a + hh * (2 * ND);
    float s = 0.f, t = 0.f;
#pragma unroll 4
    for (int d = 0; d < ND; ++d) {
        float v = bf2f(col[(size_t)d * NN]);
        s = fmaf(v, as[d], s);
        t = fmaf(v, as[ND + d], t);
    }
    srcv[idx] = s;
    tgtv[idx] = t;
}

// ---------------------------------------------------------------------------
// Flash-style attention + aggregate + mean over b.
// One wave handles 16 rows for one (t,h). Pass 1: online max/sum of
// e_ij = mask(lrelu(src_i + tgt_j)) (never materialized). Pass 2: rebuild
// normalized P per 32-col tile as bf16 A-fragment, accumulate P @ ht with
// 8 WMMAs (16x128 f32 accumulator). Both b values are summed into the same
// accumulators (mean = 0.5 * sum).
// ---------------------------------------------------------------------------
__global__ __launch_bounds__(128)
void gat_attn_kernel(const u16* __restrict__ htT, const int* __restrict__ adj,
                     const float* __restrict__ srcv, const float* __restrict__ tgtv,
                     float* __restrict__ out) {
    const int lane = threadIdx.x & 31;
    const int wave = threadIdx.x >> 5;
    const int hh   = blockIdx.y;
    const int tt   = blockIdx.z;
    const int i0   = (blockIdx.x * 4 + wave) * 16;
    const int r    = lane & 15;          // row within tile (also N index of B)
    const int half = lane >> 4;
    const int kb   = half << 3;
    const int irow = i0 + r;
    const int* arow = adj + (size_t)irow * NN;

    v8f acc[8] = {};

    for (int b = 0; b < NB; ++b) {
        const int bt = b * NT + tt;
        const float* sv    = srcv + ((size_t)bt * NH + hh) * NN;
        const float* tv    = tgtv + ((size_t)bt * NH + hh) * NN;
        const u16*   vbase = htT + (size_t)bt * (NN * FOUT) + (size_t)(hh * ND) * NN;
        const float  si    = sv[irow];

        // ---- pass 1: online softmax stats, lane halves split the columns ----
        float m = -3.0e38f, ssum = 0.f;
#pragma unroll 4
        for (int jj = 0; jj < NN / 2; ++jj) {
            int j = half * (NN / 2) + jj;
            float x = si + tv[j];
            x = (x >= 0.f) ? x : SLOPE * x;
            x = (arow[j] > 0) ? x : NEGINF;
            float mn = fmaxf(m, x);
            ssum = ssum * __expf(m - mn) + __expf(x - mn);
            m = mn;
        }
        // combine the two halves (wave32: xor-16 partner holds same row)
        float mo = __shfl_xor(m, 16, 32);
        float so = __shfl_xor(ssum, 16, 32);
        float mn = fmaxf(m, mo);
        ssum = ssum * __expf(m - mn) + so * __expf(mo - mn);
        m = mn;
        const float rinv = 1.0f / ssum;

        // ---- pass 2: rebuild P tile-by-tile, WMMA-accumulate P @ ht ----
        for (int jt = 0; jt < NN / 32; ++jt) {
            const int j0 = jt * 32;
            float tvv[16];
            int   avv[16];
            *(float4*)&tvv[0]  = *(const float4*)(tv + j0 + kb);
            *(float4*)&tvv[4]  = *(const float4*)(tv + j0 + kb + 4);
            *(float4*)&tvv[8]  = *(const float4*)(tv + j0 + 16 + kb);
            *(float4*)&tvv[12] = *(const float4*)(tv + j0 + 16 + kb + 4);
            *(int4*)&avv[0]    = *(const int4*)(arow + j0 + kb);
            *(int4*)&avv[4]    = *(const int4*)(arow + j0 + kb + 4);
            *(int4*)&avv[8]    = *(const int4*)(arow + j0 + 16 + kb);
            *(int4*)&avv[12]   = *(const int4*)(arow + j0 + 16 + kb + 4);

            Frag p;
#pragma unroll
            for (int e = 0; e < 16; ++e) {
                float x = si + tvv[e];
                x = (x >= 0.f) ? x : SLOPE * x;
                float pe = (avv[e] > 0) ? __expf(x - m) * rinv : 0.f;
                p.s[e] = f2bf(pe);
            }
#pragma unroll
            for (int dt = 0; dt < 8; ++dt) {
                const u16* bcol = vbase + (size_t)(dt * 16 + r) * NN;
                Frag bf;
                bf.q[0] = *(const uint4*)(bcol + j0 + kb);
                bf.q[1] = *(const uint4*)(bcol + j0 + 16 + kb);
                acc[dt] = __builtin_amdgcn_wmma_f32_16x16x32_bf16(
                    false, p.v, false, bf.v, (short)0, acc[dt], false, false);
            }
        }
    }

    // ---- store 16x128 tile, mean over b ----
#pragma unroll
    for (int dt = 0; dt < 8; ++dt) {
        const int d = dt * 16 + r;
#pragma unroll
        for (int v = 0; v < 8; ++v) {
            const int M = v + 8 * half;
            out[(((size_t)tt * NH + hh) * NN + i0 + M) * ND + d] = acc[dt][v] * 0.5f;
        }
    }
}

// ---------------------------------------------------------------------------
// Launch: convert -> GEMM1 (WMMA) -> scores -> flash attention (WMMA)
// ---------------------------------------------------------------------------
extern "C" void kernel_launch(void* const* d_in, const int* in_sizes, int n_in,
                              void* d_out, int out_size, void* d_ws, size_t ws_size,
                              hipStream_t stream) {
    const float* h   = (const float*)d_in[0];
    const int*   adj = (const int*)  d_in[1];
    const float* W   = (const float*)d_in[2];
    const float* a   = (const float*)d_in[3];
    float*       out = (float*)d_out;

    char* ws = (char*)d_ws;
    u16*   WT   = (u16*)ws;                                        // 512 KB
    u16*   hb   = (u16*)(ws + (1u << 20));                         // 4 MB
    u16*   htT  = (u16*)(ws + (1u << 20) + (4u << 20));            // 16 MB
    float* srcv = (float*)(ws + (1u << 20) + (4u << 20) + (16u << 20)); // 256 KB
    float* tgtv = srcv + (size_t)NB * NT * NH * NN;                // 256 KB

    const int nW = FOUT * FIN;                  // 262144
    const int nH = NB * NT * NN * FIN;          // 2097152
    cvt_w_kernel<<<(nW + 255) / 256, 256, 0, stream>>>(W, WT);
    cvt_h_kernel<<<(nH + 255) / 256, 256, 0, stream>>>(h, hb, nH);

    dim3 g1(FOUT / 128, NN / 64, NB * NT);      // (8, 16, 8)
    gat_gemm1_kernel<<<g1, 256, 0, stream>>>(hb, WT, htT);

    const int nS = NB * NT * NH * NN;           // 65536
    gat_scores_kernel<<<nS / 256, 256, 0, stream>>>(htT, a, srcv, tgtv);

    dim3 g2(NN / 64, NH, NT);                   // (16, 8, 4), 4 waves/block
    gat_attn_kernel<<<g2, 128, 0, stream>>>(htT, adj, srcv, tgtv, out);
}
